// InGram_72533407695108
// MI455X (gfx1250) — matrix operand order repaired and checked by reading, not compiled, not run on hardware
//
#include <hip/hip_runtime.h>

// ---------------- problem constants (from reference) ----------------
#define E_CNT   300000
#define ER_CNT  50000
#define NE      20000          // NUM_ENT
#define NR      500            // NUM_REL
#define MROWS   (E_CNT + NE)   // 320000 edge rows incl. self loops
#define DE      128
#define DR      64
#define HH      8
#define EPSF    1e-16f

typedef __attribute__((ext_vector_type(16))) _Float16 v16h;
typedef __attribute__((ext_vector_type(8)))  float    v8f;

__device__ __forceinline__ unsigned encf(float f) {
  unsigned u = __float_as_uint(f);
  return (u & 0x80000000u) ? ~u : (u | 0x80000000u);   // order-preserving key
}
__device__ __forceinline__ float decf(unsigned u) {
  u = (u & 0x80000000u) ? (u & 0x7fffffffu) : ~u;
  return __uint_as_float(u);
}
__device__ __forceinline__ float leakyf(float x) { return x > 0.f ? x : 0.2f * x; }

// ---------------- generic small kernels ----------------
__global__ void fill_f32_kernel(float* p, float v, int n) {
  int i = blockIdx.x * blockDim.x + threadIdx.x;
  if (i < n) p[i] = v;
}

__global__ void cvt_f16_kernel(const float* __restrict__ s, _Float16* __restrict__ d, int n) {
  int i = blockIdx.x * blockDim.x + threadIdx.x;
  if (i < n) d[i] = (_Float16)s[i];
}

// Y[m,n] = b[n] + sum_k X[m,k]*W[n,k]
__global__ void lin_kernel(const float* __restrict__ X, const float* __restrict__ W,
                           const float* __restrict__ b, float* __restrict__ Y,
                           int Mr, int N, int K) {
  int i = blockIdx.x * blockDim.x + threadIdx.x;
  if (i >= Mr * N) return;
  int m = i / N, n = i % N;
  const float* x = X + (size_t)m * K;
  const float* w = W + (size_t)n * K;
  float acc = b[n];
  for (int k = 0; k < K; ++k) acc += x[k] * w[k];
  Y[i] = acc;
}

// Y[m,n] = relu(U[m,n] + b[n] + sum_k X[m,k]*W[n,k])
__global__ void lin_res_relu_kernel(const float* __restrict__ X, const float* __restrict__ W,
                                    const float* __restrict__ b, const float* __restrict__ U,
                                    float* __restrict__ Y, int Mr, int N, int K) {
  int i = blockIdx.x * blockDim.x + threadIdx.x;
  if (i >= Mr * N) return;
  int m = i / N, n = i % N;
  const float* x = X + (size_t)m * K;
  const float* w = W + (size_t)n * K;
  float acc = b[n] + U[i];
  for (int k = 0; k < K; ++k) acc += x[k] * w[k];
  Y[i] = fmaxf(acc, 0.f);
}

// ---------------- relation layer (small: plain VALU) ----------------
__global__ void rel_raw_kernel(const float* __restrict__ xrel, const int* __restrict__ rtrip,
                               const float* __restrict__ Wa, const float* __restrict__ ba,
                               const float* __restrict__ abin, const float* __restrict__ vec,
                               float* __restrict__ rawR) {
  __shared__ float cat[2 * DR];
  __shared__ float pr[DR];
  int e = blockIdx.x, n = threadIdx.x;           // 64 threads
  int hi = rtrip[e * 3 + 0], ti = rtrip[e * 3 + 1], bn = rtrip[e * 3 + 2];
  cat[n]      = xrel[hi * DR + n];
  cat[DR + n] = xrel[ti * DR + n];
  __syncthreads();
  float z = ba[n];
  for (int k = 0; k < 2 * DR; ++k) z += cat[k] * Wa[n * (2 * DR) + k];
  z = leakyf(z);
  int h = n >> 3, d = n & 7;                     // DHR = 8
  pr[n] = z * vec[h * 8 + d];
  __syncthreads();
  if (d == 0) {
    float s = 0.f;
    for (int j = 0; j < 8; ++j) s += pr[n + j];
    rawR[e * HH + h] = s + abin[bn * HH + h];
  }
}

__global__ void rel_segmax_kernel(const float* __restrict__ raw, const int* __restrict__ rtrip,
                                  unsigned* __restrict__ mx) {
  int i = blockIdx.x * blockDim.x + threadIdx.x;
  if (i >= ER_CNT * HH) return;
  int e = i >> 3, h = i & 7;
  int hi = rtrip[e * 3 + 0];
  atomicMax(&mx[hi * HH + h], encf(raw[i]));
}

__global__ void rel_expsum_kernel(float* __restrict__ raw, const int* __restrict__ rtrip,
                                  const unsigned* __restrict__ mx, float* __restrict__ s) {
  int i = blockIdx.x * blockDim.x + threadIdx.x;
  if (i >= ER_CNT * HH) return;
  int e = i >> 3, h = i & 7;
  int hi = rtrip[e * 3 + 0];
  float a = expf(raw[i] - decf(mx[hi * HH + h]));
  raw[i] = a;                                    // raw buffer now holds 'a'
  atomicAdd(&s[hi * HH + h], a);
}

__global__ void rel_aggr_kernel(const float* __restrict__ xrel, const int* __restrict__ rtrip,
                                const float* __restrict__ Wg, const float* __restrict__ bg,
                                const float* __restrict__ aR, const float* __restrict__ sR,
                                float* __restrict__ upd) {
  __shared__ float xt[DR];
  int e = blockIdx.x, n = threadIdx.x;           // 64 threads
  int hi = rtrip[e * 3 + 0], ti = rtrip[e * 3 + 1];
  xt[n] = xrel[ti * DR + n];
  __syncthreads();
  float g = bg[n];
  for (int k = 0; k < DR; ++k) g += xt[k] * Wg[n * DR + k];
  int h = n >> 3;
  float beta = aR[e * HH + h] / (sR[hi * HH + h] + EPSF);
  atomicAdd(&upd[hi * DR + n], beta * g);
}

// ---------------- entity self-relation prep ----------------
__global__ void selfrel_accum_kernel(const int* __restrict__ trip, const float* __restrict__ xrel,
                                     float* __restrict__ selfrel, float* __restrict__ freq) {
  int i = blockIdx.x * blockDim.x + threadIdx.x;
  if (i >= E_CNT * DR) return;
  int e = i >> 6, d = i & 63;
  int r = trip[e * 3 + 1], t = trip[e * 3 + 2];
  atomicAdd(&selfrel[t * DR + d], xrel[r * DR + d]);
  if (d == 0) atomicAdd(&freq[t], 1.0f);
}

__global__ void selfrel_norm_kernel(float* __restrict__ selfrel, const float* __restrict__ freq) {
  int i = blockIdx.x * blockDim.x + threadIdx.x;
  if (i >= NE * DR) return;
  selfrel[i] = selfrel[i] / (freq[i >> 6] + EPSF);
}

// ---------------- entity layer: WMMA attention logits ----------------
// Tile = 16 edges; 8 waves per block, wave w = head h. K = 320, 10 wmma steps.
#define LDA_ATT 328
__global__ __launch_bounds__(256) void ent_attn_kernel(
    const float* __restrict__ xent, const float* __restrict__ xrel,
    const float* __restrict__ selfrel, const int* __restrict__ trip,
    const _Float16* __restrict__ WaH, const float* __restrict__ ba,
    const float* __restrict__ vec, float* __restrict__ rawE) {
  __shared__ _Float16 lA[16 * LDA_ATT];
  int tid = threadIdx.x, tile = blockIdx.x;
  // ---- gather phase: 16 threads per edge row, 20 features each (320 total)
  int row = tid >> 4, part = tid & 15;
  int e = tile * 16 + row;
  bool valid = (e < MROWS);
  const float* pT = xent; const float* pH = xent; const float* pR = xrel;
  if (valid) {
    int head, tail;
    if (e < E_CNT) {
      head = trip[e * 3 + 0]; tail = trip[e * 3 + 2];
      pR = xrel + trip[e * 3 + 1] * DR;
    } else {
      head = tail = e - E_CNT;
      pR = selfrel + (e - E_CNT) * DR;
    }
    pT = xent + (size_t)tail * DE;
    pH = xent + (size_t)head * DE;
  }
#pragma unroll
  for (int j = 0; j < 20; ++j) {
    int c = part * 20 + j;
    float fv = 0.f;
    if (valid) fv = (c < 128) ? pT[c] : ((c < 256) ? pH[c - 128] : pR[c - 256]);
    lA[row * LDA_ATT + c] = (_Float16)fv;
  }
  __syncthreads();
  // ---- WMMA phase
  int w = tid >> 5;                  // head index 0..7
  int lane = tid & 31;
  int d_idx = lane & 15, hf = lane >> 4;
  int n_g = w * 16 + d_idx;          // global output column 0..127
  const _Float16* Arow = lA + (lane & 15) * LDA_ATT;
  const _Float16* Brow = WaH + (size_t)n_g * 320;  // B[k][n] = Wa[n][k]
  v8f acc = {};
#pragma unroll
  for (int kk = 0; kk < 320; kk += 32) {
    v16h af, bf;
#pragma unroll
    for (int v = 0; v < 8; ++v) {
      int ka = (v < 4) ? (2 * v + 8 * hf) : (16 + 2 * (v - 4) + 8 * hf);
      af[2 * v]     = Arow[kk + ka];
      af[2 * v + 1] = Arow[kk + ka + 1];
      int kb = (v < 4) ? (2 * v + 8 * hf) : (16 + 2 * (v - 4) + 8 * hf);
      bf[2 * v]     = Brow[kk + kb];
      bf[2 * v + 1] = Brow[kk + kb + 1];
    }
    acc = __builtin_amdgcn_wmma_f32_16x16x32_f16(false, af, false, bf, (short)0, acc,
                                                 false, false);
  }
  // ---- epilogue: leaky, dot with vec over the 16 head dims (cross-lane)
  float bb = ba[n_g];
  float vv = vec[w * 16 + d_idx];
  float red[8];
#pragma unroll
  for (int v = 0; v < 8; ++v) red[v] = leakyf(acc[v] + bb) * vv;
#pragma unroll
  for (int off = 8; off >= 1; off >>= 1) {
#pragma unroll
    for (int v = 0; v < 8; ++v) red[v] += __shfl_xor(red[v], off, 16);
  }
  if (d_idx == 0) {
#pragma unroll
    for (int v = 0; v < 8; ++v) {
      int rr = v + 8 * hf;
      int ee = tile * 16 + rr;
      if (ee < MROWS) rawE[ee * HH + w] = red[v];
    }
  }
}

// ---------------- entity segment softmax pieces ----------------
__global__ void ent_segmax_kernel(const float* __restrict__ raw, const int* __restrict__ trip,
                                  unsigned* __restrict__ mx) {
  int i = blockIdx.x * blockDim.x + threadIdx.x;
  if (i >= MROWS * HH) return;
  int e = i >> 3, h = i & 7;
  int tail = (e < E_CNT) ? trip[e * 3 + 2] : (e - E_CNT);
  atomicMax(&mx[tail * HH + h], encf(raw[i]));
}

__global__ void ent_expsum_kernel(float* __restrict__ raw, const int* __restrict__ trip,
                                  const unsigned* __restrict__ mx, float* __restrict__ s) {
  int i = blockIdx.x * blockDim.x + threadIdx.x;
  if (i >= MROWS * HH) return;
  int e = i >> 3, h = i & 7;
  int tail = (e < E_CNT) ? trip[e * 3 + 2] : (e - E_CNT);
  float a = expf(raw[i] - decf(mx[tail * HH + h]));
  raw[i] = a;
  atomicAdd(&s[tail * HH + h], a);
}

// ---------------- entity layer: WMMA aggregation + segment sum ----------------
#define LDA_AGG 200
__global__ __launch_bounds__(256) void ent_aggr_kernel(
    const float* __restrict__ xent, const float* __restrict__ xrel,
    const float* __restrict__ selfrel, const int* __restrict__ trip,
    const _Float16* __restrict__ WgH, const float* __restrict__ bg,
    const float* __restrict__ aE, const float* __restrict__ sE,
    float* __restrict__ upd) {
  __shared__ _Float16 lA[16 * LDA_AGG];
  __shared__ float lbeta[16 * HH];
  __shared__ int   ltail[16];
  int tid = threadIdx.x, tile = blockIdx.x;
  int row = tid >> 4, part = tid & 15;
  int e = tile * 16 + row;
  bool valid = (e < MROWS);
  const float* pH = xent; const float* pR = xrel;
  int tail = 0;
  if (valid) {
    int head;
    if (e < E_CNT) {
      head = trip[e * 3 + 0]; tail = trip[e * 3 + 2];
      pR = xrel + trip[e * 3 + 1] * DR;
    } else {
      head = tail = e - E_CNT;
      pR = selfrel + (e - E_CNT) * DR;
    }
    pH = xent + (size_t)head * DE;
  }
#pragma unroll
  for (int j = 0; j < 12; ++j) {
    int c = part * 12 + j;
    float fv = 0.f;
    if (valid) fv = (c < 128) ? pH[c] : pR[c - 128];
    lA[row * LDA_AGG + c] = (_Float16)fv;
  }
  if (part == 0) ltail[row] = valid ? tail : 0;
  __syncthreads();
  if (tid < 128) {
    int rr = tid >> 3, h = tid & 7;
    int ee = tile * 16 + rr;
    lbeta[tid] = (ee < MROWS)
                     ? aE[ee * HH + h] / (sE[ltail[rr] * HH + h] + EPSF)
                     : 0.f;
  }
  __syncthreads();
  int w = tid >> 5;
  int lane = tid & 31;
  int d_idx = lane & 15, hf = lane >> 4;
  int n_g = w * 16 + d_idx;
  const _Float16* Arow = lA + (lane & 15) * LDA_AGG;
  const _Float16* Brow = WgH + (size_t)n_g * 192;
  v8f acc = {};
#pragma unroll
  for (int kk = 0; kk < 192; kk += 32) {
    v16h af, bf;
#pragma unroll
    for (int v = 0; v < 8; ++v) {
      int ka = (v < 4) ? (2 * v + 8 * hf) : (16 + 2 * (v - 4) + 8 * hf);
      af[2 * v]     = Arow[kk + ka];
      af[2 * v + 1] = Arow[kk + ka + 1];
      int kb = (v < 4) ? (2 * v + 8 * hf) : (16 + 2 * (v - 4) + 8 * hf);
      bf[2 * v]     = Brow[kk + kb];
      bf[2 * v + 1] = Brow[kk + kb + 1];
    }
    acc = __builtin_amdgcn_wmma_f32_16x16x32_f16(false, af, false, bf, (short)0, acc,
                                                 false, false);
  }
  float bgv = bg[n_g];
#pragma unroll
  for (int v = 0; v < 8; ++v) {
    int rr = v + 8 * hf;
    float val = (acc[v] + bgv) * lbeta[rr * HH + w];
    atomicAdd(&upd[(size_t)ltail[rr] * DE + n_g], val);  // pad rows add 0.0
  }
}

// ---------------- host-side orchestration ----------------
#define CDIV(a, b) (((a) + (b) - 1) / (b))

extern "C" void kernel_launch(void* const* d_in, const int* in_sizes, int n_in,
                              void* d_out, int out_size, void* d_ws, size_t ws_size,
                              hipStream_t stream) {
  (void)in_sizes; (void)n_in; (void)out_size; (void)ws_size;
  const float* emb_ent = (const float*)d_in[0];
  const float* emb_rel = (const float*)d_in[1];
  const int*   trip    = (const int*)d_in[2];
  const int*   rtrip   = (const int*)d_in[3];
  const float* p1eW = (const float*)d_in[4];
  const float* p1eB = (const float*)d_in[5];
  const float* p1rW = (const float*)d_in[6];
  const float* p1rB = (const float*)d_in[7];
  const float* raW  = (const float*)d_in[8];
  const float* raB  = (const float*)d_in[9];
  const float* raBin= (const float*)d_in[10];
  const float* raVec= (const float*)d_in[11];
  const float* rgW  = (const float*)d_in[12];
  const float* rgB  = (const float*)d_in[13];
  const float* rrW  = (const float*)d_in[14];
  const float* rrB  = (const float*)d_in[15];
  const float* eaW  = (const float*)d_in[16];
  const float* eaB  = (const float*)d_in[17];
  const float* eaVec= (const float*)d_in[18];
  const float* egW  = (const float*)d_in[19];
  const float* egB  = (const float*)d_in[20];
  const float* reW  = (const float*)d_in[21];
  const float* reB  = (const float*)d_in[22];
  const float* p2eW = (const float*)d_in[23];
  const float* p2eB = (const float*)d_in[24];
  const float* p2rW = (const float*)d_in[25];
  const float* p2rB = (const float*)d_in[26];
  float* out = (float*)d_out;

  char* ws = (char*)d_ws;
  size_t off = 0;
  auto take = [&](size_t bytes) -> char* {
    char* p = ws + off;
    off += (bytes + 255) & ~(size_t)255;
    return p;
  };
  float* xent0   = (float*)take((size_t)NE * DE * 4);
  float* xent1   = (float*)take((size_t)NE * DE * 4);
  float* updE    = (float*)take((size_t)NE * DE * 4);
  float* xrel0   = (float*)take((size_t)NR * DR * 4);
  float* xrel1   = (float*)take((size_t)NR * DR * 4);
  float* updR    = (float*)take((size_t)NR * DR * 4);
  float* selfrel = (float*)take((size_t)NE * DR * 4);
  float* freq    = (float*)take((size_t)NE * 4);
  float* rawE    = (float*)take((size_t)MROWS * HH * 4);
  unsigned* mxE  = (unsigned*)take((size_t)NE * HH * 4);
  float* sE      = (float*)take((size_t)NE * HH * 4);
  float* rawR    = (float*)take((size_t)ER_CNT * HH * 4);
  unsigned* mxR  = (unsigned*)take((size_t)NR * HH * 4);
  float* sR      = (float*)take((size_t)NR * HH * 4);
  _Float16* wAttnH = (_Float16*)take((size_t)2 * DE * 320 * 2);
  _Float16* wAggrH = (_Float16*)take((size_t)2 * DE * 192 * 2);

  // f16 copies of the big edge-GEMM weights
  cvt_f16_kernel<<<CDIV(2 * DE * 320, 256), 256, 0, stream>>>(eaW, wAttnH, 2 * DE * 320);
  cvt_f16_kernel<<<CDIV(2 * DE * 192, 256), 256, 0, stream>>>(egW, wAggrH, 2 * DE * 192);

  // input projections
  lin_kernel<<<CDIV(NE * DE, 256), 256, 0, stream>>>(emb_ent, p1eW, p1eB, xent0, NE, DE, 32);
  lin_kernel<<<CDIV(NR * DR, 256), 256, 0, stream>>>(emb_rel, p1rW, p1rB, xrel0, NR, DR, 16);

  // ---- relation layers ----
  for (int l = 0; l < 2; ++l) {
    const float* xin = l ? xrel1 : xrel0;
    float* xout      = l ? xrel0 : xrel1;
    fill_f32_kernel<<<CDIV(NR * HH, 256), 256, 0, stream>>>((float*)mxR, 0.f, NR * HH);
    fill_f32_kernel<<<CDIV(NR * HH, 256), 256, 0, stream>>>(sR, 0.f, NR * HH);
    fill_f32_kernel<<<CDIV(NR * DR, 256), 256, 0, stream>>>(updR, 0.f, NR * DR);
    rel_raw_kernel<<<ER_CNT, 64, 0, stream>>>(xin, rtrip, raW + l * DR * 2 * DR,
                                              raB + l * DR, raBin + l * 10 * HH,
                                              raVec + l * HH * 8, rawR);
    rel_segmax_kernel<<<CDIV(ER_CNT * HH, 256), 256, 0, stream>>>(rawR, rtrip, mxR);
    rel_expsum_kernel<<<CDIV(ER_CNT * HH, 256), 256, 0, stream>>>(rawR, rtrip, mxR, sR);
    rel_aggr_kernel<<<ER_CNT, 64, 0, stream>>>(xin, rtrip, rgW + l * DR * DR,
                                               rgB + l * DR, rawR, sR, updR);
    lin_res_relu_kernel<<<CDIV(NR * DR, 256), 256, 0, stream>>>(
        xin, rrW + l * DR * DR, rrB + l * DR, updR, xout, NR, DR, DR);
  }
  // final x_rel is in xrel0

  // ---- self-relation features (fixed across both entity layers) ----
  fill_f32_kernel<<<CDIV(NE, 256), 256, 0, stream>>>(freq, 0.f, NE);
  fill_f32_kernel<<<CDIV(NE * DR, 256), 256, 0, stream>>>(selfrel, 0.f, NE * DR);
  selfrel_accum_kernel<<<CDIV(E_CNT * DR, 256), 256, 0, stream>>>(trip, xrel0, selfrel, freq);
  selfrel_norm_kernel<<<CDIV(NE * DR, 256), 256, 0, stream>>>(selfrel, freq);

  // ---- entity layers (WMMA) ----
  const int ntiles = CDIV(MROWS, 16);
  for (int l = 0; l < 2; ++l) {
    const float* xin = l ? xent1 : xent0;
    float* xout      = l ? xent0 : xent1;
    fill_f32_kernel<<<CDIV(NE * HH, 256), 256, 0, stream>>>((float*)mxE, 0.f, NE * HH);
    fill_f32_kernel<<<CDIV(NE * HH, 256), 256, 0, stream>>>(sE, 0.f, NE * HH);
    fill_f32_kernel<<<CDIV(NE * DE, 256), 256, 0, stream>>>(updE, 0.f, NE * DE);
    ent_attn_kernel<<<ntiles, 256, 0, stream>>>(xin, xrel0, selfrel, trip,
                                                wAttnH + (size_t)l * DE * 320,
                                                eaB + l * DE, eaVec + l * HH * 16, rawE);
    ent_segmax_kernel<<<CDIV(MROWS * HH, 256), 256, 0, stream>>>(rawE, trip, mxE);
    ent_expsum_kernel<<<CDIV(MROWS * HH, 256), 256, 0, stream>>>(rawE, trip, mxE, sE);
    ent_aggr_kernel<<<ntiles, 256, 0, stream>>>(xin, xrel0, selfrel, trip,
                                                wAggrH + (size_t)l * DE * 192,
                                                egB + l * DE, rawE, sE, updE);
    lin_res_relu_kernel<<<CDIV(NE * DE, 256), 256, 0, stream>>>(
        xin, reW + l * DE * DE, reB + l * DE, updE, xout, NE, DE, DE);
  }
  // final x_ent is in xent0

  // ---- output projections (concatenated: ent then rel) ----
  lin_kernel<<<CDIV(NE * 32, 256), 256, 0, stream>>>(xent0, p2eW, p2eB, out, NE, 32, DE);
  lin_kernel<<<CDIV(NR * 16, 256), 256, 0, stream>>>(xrel0, p2rW, p2rB,
                                                     out + (size_t)NE * 32, NR, 16, DR);
}
